// KVCacheMHA_39144331936037
// MI455X (gfx1250) — matrix operand order repaired and checked by reading, not compiled, not run on hardware
//
#include <hip/hip_runtime.h>

// ---------------------------------------------------------------------------
// KV-cache MHA for MI455X (gfx1250), wave32, WMMA bf16 16x16x32 everywhere.
// B=2, S=2048, D=2048, H=16, HD=128, PAST=2048, k_len=4096.
// ~275 GFLOP vs ~17us of HBM traffic -> compute bound -> everything on
// v_wmma_f32_16x16x32_bf16 (f32 accumulate), staged via async-to-LDS with
// ASYNCcnt-pipelined double buffering where the HW path exists.
// ---------------------------------------------------------------------------

#define DK 2048          // model dim (= GEMM K and N)
#define MROWS 4096       // B*S
#define PAST 2048
#define KLEN 4096
#define HEADS 16
#define HD 128

typedef float v8f  __attribute__((ext_vector_type(8)));
typedef __bf16 v16bf __attribute__((ext_vector_type(16)));
typedef int    gint4 __attribute__((vector_size(16)));   // matches builtin proto

union BF16x16 { uint4 q[2]; v16bf v; };   // 32 bytes: one 16-bf16 fragment

#if __has_builtin(__builtin_amdgcn_global_load_async_to_lds_b128) && \
    __has_builtin(__builtin_amdgcn_s_wait_asynccnt)
#define USE_ASYNC_LDS 1
// global_load_async_to_lds_b128: (global int4* src, lds int4* dst, imm off, imm cpol)
#define ASYNC_CP16(dst, src)                                            \
  __builtin_amdgcn_global_load_async_to_lds_b128(                       \
      (__attribute__((address_space(1))) gint4*)(src),                  \
      (__attribute__((address_space(3))) gint4*)(dst), 0, 0)
#else
#define USE_ASYNC_LDS 0
#endif

// ---------------------------------------------------------------------------
// x (f32) -> bf16, vectorized float4 -> packed 4x bf16 (uint2)
// ---------------------------------------------------------------------------
__global__ void cvt_bf16_kernel(const float* __restrict__ src,
                                __bf16* __restrict__ dst, int n4) {
  int i = blockIdx.x * blockDim.x + threadIdx.x;
  if (i < n4) {
    float4 f = ((const float4*)src)[i];
    union { __bf16 h[4]; uint2 u; } p;
    p.h[0] = (__bf16)f.x; p.h[1] = (__bf16)f.y;
    p.h[2] = (__bf16)f.z; p.h[3] = (__bf16)f.w;
    ((uint2*)dst)[i] = p.u;
  }
}

// ---------------------------------------------------------------------------
// W [K][N] f32 -> Wt [N][K] bf16 (LDS-tiled transpose, coalesced both sides)
// ---------------------------------------------------------------------------
__global__ __launch_bounds__(256) void transpose_w_kernel(
    const float* __restrict__ W, __bf16* __restrict__ Wt) {
  __shared__ float tile[32][33];
  const int tx = threadIdx.x, ty = threadIdx.y;      // block (32,8)
  const int n0 = blockIdx.x * 32, k0 = blockIdx.y * 32;
  for (int j = 0; j < 32; j += 8)
    tile[ty + j][tx] = W[(size_t)(k0 + ty + j) * DK + n0 + tx];
  __syncthreads();
  for (int j = 0; j < 32; j += 8)
    Wt[(size_t)(n0 + ty + j) * DK + k0 + tx] = (__bf16)tile[tx][ty + j];
}

// ---------------------------------------------------------------------------
// past_k/past_v [B,H,PAST,HD] -> K/V out [B,H,KLEN,HD] (rows 0..PAST-1)
// ---------------------------------------------------------------------------
__global__ void copy_past_kernel(const float* __restrict__ pk,
                                 const float* __restrict__ pv,
                                 float* __restrict__ Kout,
                                 float* __restrict__ Vout, int n4) {
  int i = blockIdx.x * blockDim.x + threadIdx.x;
  if (i < n4) {
    const int chunk = PAST * HD / 4;                 // float4 per (b,h)
    int bh = i / chunk, r = i % chunk;
    ((float4*)Kout)[(size_t)bh * (KLEN * HD / 4) + r] = ((const float4*)pk)[i];
    ((float4*)Vout)[(size_t)bh * (KLEN * HD / 4) + r] = ((const float4*)pv)[i];
  }
}

// ---------------------------------------------------------------------------
// C[M=4096][N=2048] = A_bf16 @ Wt_bf16^T + bias.  128x128 tile / workgroup,
// K-step 32, 8 waves each doing a 32x64 subtile (2x4 WMMA frags).
// Async path: double-buffered LDS, one barrier per K-step; next tile's
// global_load_async_to_lds_b128 issued right after the barrier so HBM latency
// overlaps the 8 WMMAs. Fallback: register-staged single buffer.
// mode 0: bf16 row-major out; mode 1: f32 -> [b,h,PAST+q,hd]; mode 2: f32 rm.
// ---------------------------------------------------------------------------
__global__ __launch_bounds__(256) void gemm_wmma_kernel(
    const __bf16* __restrict__ A, const __bf16* __restrict__ Bt,
    const float* __restrict__ bias, float* __restrict__ outF,
    __bf16* __restrict__ outH, int mode) {
#if USE_ASYNC_LDS
  __shared__ __bf16 As[2][128][40];  // 40-elem stride: 80B = 16B-aligned, conflict-free
  __shared__ __bf16 Bs[2][128][40];
#else
  __shared__ __bf16 As[1][128][40];
  __shared__ __bf16 Bs[1][128][40];
#endif

  const int t = threadIdx.x;
  const int lane = t & 31, w = t >> 5;
  const int wr = w >> 1, wc = w & 1;           // 4x2 wave grid
  const int hl = lane >> 4, c0 = lane & 15;
  const int m0 = blockIdx.y * 128, n0 = blockIdx.x * 128;
  const int srow = t >> 1, shalf = t & 1;      // staging: 128 rows x 2 halves

  const __bf16* gA = A  + (size_t)(m0 + srow) * DK + shalf * 16;
  const __bf16* gB = Bt + (size_t)(n0 + srow) * DK + shalf * 16;

  v8f acc[2][4];
  for (int i = 0; i < 2; i++)
    for (int j = 0; j < 4; j++)
      for (int r = 0; r < 8; r++) acc[i][j][r] = 0.f;

#if USE_ASYNC_LDS
  // prologue: tile 0 -> buffer 0
  ASYNC_CP16(&As[0][srow][shalf * 16],     gA);
  ASYNC_CP16(&As[0][srow][shalf * 16 + 8], gA + 8);
  ASYNC_CP16(&Bs[0][srow][shalf * 16],     gB);
  ASYNC_CP16(&Bs[0][srow][shalf * 16 + 8], gB + 8);
#else
  uint4 ra0, ra1, rb0, rb1;
  ra0 = ((const uint4*)gA)[0]; ra1 = ((const uint4*)gA)[1];
  rb0 = ((const uint4*)gB)[0]; rb1 = ((const uint4*)gB)[1];
#endif

  for (int it = 0; it < DK / 32; ++it) {
#if USE_ASYNC_LDS
    const int cur = it & 1;
    __builtin_amdgcn_s_wait_asynccnt(0);   // own async writes for tile `it` landed
    __syncthreads();                       // everyone's landed; prev compute done
    if (it + 1 < DK / 32) {                // issue next tile into other buffer
      const int kn = (it + 1) * 32;
      ASYNC_CP16(&As[cur ^ 1][srow][shalf * 16],     gA + kn);
      ASYNC_CP16(&As[cur ^ 1][srow][shalf * 16 + 8], gA + kn + 8);
      ASYNC_CP16(&Bs[cur ^ 1][srow][shalf * 16],     gB + kn);
      ASYNC_CP16(&Bs[cur ^ 1][srow][shalf * 16 + 8], gB + kn + 8);
    }
#else
    const int cur = 0;
    __syncthreads();                       // LDS free
    {
      uint4* la = (uint4*)&As[0][srow][shalf * 16];
      la[0] = ra0; la[1] = ra1;
      uint4* lb = (uint4*)&Bs[0][srow][shalf * 16];
      lb[0] = rb0; lb[1] = rb1;
      if (it + 1 < DK / 32) {
        const int kn = (it + 1) * 32;
        ra0 = ((const uint4*)(gA + kn))[0]; ra1 = ((const uint4*)(gA + kn))[1];
        rb0 = ((const uint4*)(gB + kn))[0]; rb1 = ((const uint4*)(gB + kn))[1];
        __builtin_prefetch(gA + kn + 32, 0, 1);
        __builtin_prefetch(gB + kn + 32, 0, 1);
      }
    }
    __syncthreads();                       // LDS ready
#endif

    BF16x16 af[2], bf[4];
    for (int mt = 0; mt < 2; mt++) {       // A frag: 16-bit 16x32 layout
      const __bf16* ap = &As[cur][wr * 32 + mt * 16 + c0][hl * 8];
      af[mt].q[0] = *(const uint4*)ap;
      af[mt].q[1] = *(const uint4*)(ap + 16);
    }
    for (int nt = 0; nt < 4; nt++) {       // B frag: 16-bit 32x16 layout (same K order as A)
      const __bf16* bp = &Bs[cur][wc * 64 + nt * 16 + c0][hl * 8];
      bf[nt].q[0] = *(const uint4*)bp;
      bf[nt].q[1] = *(const uint4*)(bp + 16);
    }
    for (int mt = 0; mt < 2; mt++)
      for (int nt = 0; nt < 4; nt++)
        acc[mt][nt] = __builtin_amdgcn_wmma_f32_16x16x32_bf16(
            false, af[mt].v, false, bf[nt].v, (short)0, acc[mt][nt], false, false);
  }

  for (int mt = 0; mt < 2; mt++)
    for (int nt = 0; nt < 4; nt++) {
      const int gn = n0 + wc * 64 + nt * 16 + c0;
      const float bv = bias[gn];
      for (int r = 0; r < 8; r++) {
        const int gm = m0 + wr * 32 + mt * 16 + r + 8 * hl;
        const float val = acc[mt][nt][r] + bv;
        if (mode == 0) {
          outH[(size_t)gm * DK + gn] = (__bf16)val;
        } else if (mode == 1) {
          const size_t off =
              (((size_t)((gm >> 11) * HEADS + (gn >> 7))) * KLEN + PAST + (gm & 2047)) * HD +
              (gn & 127);
          outF[off] = val;
        } else {
          outF[(size_t)gm * DK + gn] = val;
        }
      }
    }
}

// ---------------------------------------------------------------------------
// Flash attention: one workgroup = (b, h, 128 q-rows); each wave owns 16 rows.
// K/V key-blocks are register-staged (f32->bf16 convert + V transpose on the
// LDS store), with block i+1's global loads issued between the barriers of
// block i so HBM latency overlaps the 16 WMMAs + softmax.
// ---------------------------------------------------------------------------
__global__ __launch_bounds__(256) void attn_wmma_kernel(
    const __bf16* __restrict__ Qb,   // [4096][2048] bf16, merged heads
    const float* __restrict__ Kf,    // [B,H,KLEN,HD] f32
    const float* __restrict__ Vf,    // [B,H,KLEN,HD] f32
    __bf16* __restrict__ Cb) {       // ctx [4096][2048] bf16
  __shared__ __bf16 Ks[32][136];     // [key][dim] : B-frag of Q*K^T
  __shared__ __bf16 Vs[128][40];     // [dim][key] : B-frag of P*V
  __shared__ __bf16 Ps[8][16][40];   // per-wave P re-fragmentation buffer

  const int t = threadIdx.x, lane = t & 31, w = t >> 5;
  const int hl = lane >> 4, c0 = lane & 15;
  const int q0 = blockIdx.x * 128, h = blockIdx.y, b = blockIdx.z;
  const int qrow = q0 + w * 16;
  const size_t kvbase = ((size_t)(b * HEADS + h)) * KLEN * HD;

  // Preload Q fragments (4 K-steps over HD=128)
  BF16x16 qf[4];
  for (int f = 0; f < 4; f++) {
    const __bf16* qp =
        Qb + (size_t)(b * 2048 + qrow + c0) * DK + h * HD + f * 32 + hl * 8;
    qf[f].q[0] = *(const uint4*)qp;
    qf[f].q[1] = *(const uint4*)(qp + 16);
  }

  float m[8], l[8];
  v8f acc[8];
  for (int r = 0; r < 8; r++) { m[r] = -1e30f; l[r] = 0.f; }
  for (int nt = 0; nt < 8; nt++)
    for (int r = 0; r < 8; r++) acc[nt][r] = 0.f;

  const int skey = t >> 3, sd = (t & 7) * 16;   // staging split: key x 16 dims
  const int nblk = q0 / 32 + 68;                // keys up to q0+127+PAST
  const float sc = 0.08838834764831845f;        // 1/sqrt(128)

  float4 kr[4], vr[4];
  {
    const float4* kp = (const float4*)(Kf + kvbase + (size_t)skey * HD + sd);
    const float4* vp = (const float4*)(Vf + kvbase + (size_t)skey * HD + sd);
    for (int i = 0; i < 4; i++) { kr[i] = kp[i]; vr[i] = vp[i]; }
  }

  for (int blk = 0; blk < nblk; blk++) {
    const int kb = blk * 32;
    __syncthreads();                        // LDS free (prev block's reads done)
    for (int i = 0; i < 4; i++) {
      const int d = sd + i * 4;
      Ks[skey][d + 0] = (__bf16)kr[i].x; Ks[skey][d + 1] = (__bf16)kr[i].y;
      Ks[skey][d + 2] = (__bf16)kr[i].z; Ks[skey][d + 3] = (__bf16)kr[i].w;
      Vs[d + 0][skey] = (__bf16)vr[i].x; Vs[d + 1][skey] = (__bf16)vr[i].y;
      Vs[d + 2][skey] = (__bf16)vr[i].z; Vs[d + 3][skey] = (__bf16)vr[i].w;
    }
    if (blk + 1 < nblk) {                   // issue next block's loads now
      const float4* kp =
          (const float4*)(Kf + kvbase + (size_t)(kb + 32 + skey) * HD + sd);
      const float4* vp =
          (const float4*)(Vf + kvbase + (size_t)(kb + 32 + skey) * HD + sd);
      for (int i = 0; i < 4; i++) { kr[i] = kp[i]; vr[i] = vp[i]; }
    }
    __syncthreads();                        // LDS ready

    // scores: two 16x16 tiles (keys kb..kb+15, kb+16..kb+31), K=128 over 4 WMMAs
    v8f s0, s1;
    for (int r = 0; r < 8; r++) { s0[r] = 0.f; s1[r] = 0.f; }
    for (int d = 0; d < 4; d++) {
      BF16x16 bk0, bk1;
      const __bf16* kp0 = &Ks[c0][d * 32 + hl * 8];
      bk0.q[0] = *(const uint4*)kp0; bk0.q[1] = *(const uint4*)(kp0 + 16);
      const __bf16* kp1 = &Ks[16 + c0][d * 32 + hl * 8];
      bk1.q[0] = *(const uint4*)kp1; bk1.q[1] = *(const uint4*)(kp1 + 16);
      s0 = __builtin_amdgcn_wmma_f32_16x16x32_bf16(false, qf[d].v, false, bk0.v,
                                                   (short)0, s0, false, false);
      s1 = __builtin_amdgcn_wmma_f32_16x16x32_bf16(false, qf[d].v, false, bk1.v,
                                                   (short)0, s1, false, false);
    }

    // online softmax update (row = r + 8*hl, its 16 cols live across a 16-lane half)
    float p0v[8], p1v[8], alf[8];
    for (int r = 0; r < 8; r++) {
      const int qq = qrow + r + 8 * hl;
      const int jlim = qq + (KLEN - 2048);      // causal: j <= q + (k_len - q_len)
      float v0 = (kb + c0 <= jlim) ? s0[r] * sc : -1e30f;
      float v1 = (kb + 16 + c0 <= jlim) ? s1[r] * sc : -1e30f;
      float mx = fmaxf(v0, v1);
      for (int off = 8; off > 0; off >>= 1) mx = fmaxf(mx, __shfl_xor(mx, off, 16));
      const float mn = fmaxf(m[r], mx);
      const float al = __expf(m[r] - mn);
      const float e0 = __expf(v0 - mn), e1 = __expf(v1 - mn);
      float rs = e0 + e1;
      for (int off = 8; off > 0; off >>= 1) rs += __shfl_xor(rs, off, 16);
      l[r] = l[r] * al + rs;
      m[r] = mn;
      alf[r] = al;
      p0v[r] = e0; p1v[r] = e1;
    }

    // re-fragment P: C-layout -> A-layout via per-wave LDS scratch
    for (int r = 0; r < 8; r++) {
      Ps[w][r + 8 * hl][c0]      = (__bf16)p0v[r];
      Ps[w][r + 8 * hl][16 + c0] = (__bf16)p1v[r];
    }
    asm volatile("s_wait_dscnt 0" ::: "memory");
    BF16x16 pf;
    const __bf16* pp = &Ps[w][c0][hl * 8];
    pf.q[0] = *(const uint4*)pp;
    pf.q[1] = *(const uint4*)(pp + 16);

    // ctx = ctx*alpha + P @ V  (8 N-tiles over HD=128)
    for (int nt = 0; nt < 8; nt++) {
      for (int r = 0; r < 8; r++) acc[nt][r] *= alf[r];
      BF16x16 vfr;
      const __bf16* vp2 = &Vs[nt * 16 + c0][hl * 8];
      vfr.q[0] = *(const uint4*)vp2; vfr.q[1] = *(const uint4*)(vp2 + 16);
      acc[nt] = __builtin_amdgcn_wmma_f32_16x16x32_bf16(
          false, pf.v, false, vfr.v, (short)0, acc[nt], false, false);
    }
  }

  for (int r = 0; r < 8; r++) {
    const float inv = 1.0f / l[r];
    const int g = b * 2048 + qrow + r + 8 * hl;
    for (int nt = 0; nt < 8; nt++)
      Cb[(size_t)g * DK + h * HD + nt * 16 + c0] = (__bf16)(acc[nt][r] * inv);
  }
}

// ---------------------------------------------------------------------------
// Launch
// ---------------------------------------------------------------------------
extern "C" void kernel_launch(void* const* d_in, const int* in_sizes, int n_in,
                              void* d_out, int out_size, void* d_ws, size_t ws_size,
                              hipStream_t stream) {
  const float* x      = (const float*)d_in[0];
  const float* past_k = (const float*)d_in[1];
  const float* past_v = (const float*)d_in[2];
  const float* Wq = (const float*)d_in[3];  const float* bq = (const float*)d_in[4];
  const float* Wk = (const float*)d_in[5];  const float* bk = (const float*)d_in[6];
  const float* Wv = (const float*)d_in[7];  const float* bv = (const float*)d_in[8];
  const float* Wo = (const float*)d_in[9];  const float* bo = (const float*)d_in[10];

  float* out  = (float*)d_out;                         // [2,2048,2048]
  float* Kout = out + (size_t)8388608;                 // [2,16,4096,128]
  float* Vout = Kout + (size_t)16777216;               // [2,16,4096,128]

  char* ws = (char*)d_ws;
  __bf16* xb  = (__bf16*)ws;                                   // 16 MiB
  __bf16* Wtq = (__bf16*)(ws + (size_t)16777216);
  __bf16* Wtk = Wtq + (size_t)4194304;
  __bf16* Wtv = Wtk + (size_t)4194304;
  __bf16* Wto = Wtv + (size_t)4194304;
  __bf16* Qb  = Wto + (size_t)4194304;                         // [4096][2048]
  __bf16* Cb  = Qb  + (size_t)8388608;                         // [4096][2048]

  // 1. convert x to bf16
  cvt_bf16_kernel<<<8192, 256, 0, stream>>>(x, xb, MROWS * DK / 4);

  // 2. transpose + convert weights
  dim3 tb(32, 8), tg(DK / 32, DK / 32);
  transpose_w_kernel<<<tg, tb, 0, stream>>>(Wq, Wtq);
  transpose_w_kernel<<<tg, tb, 0, stream>>>(Wk, Wtk);
  transpose_w_kernel<<<tg, tb, 0, stream>>>(Wv, Wtv);
  transpose_w_kernel<<<tg, tb, 0, stream>>>(Wo, Wto);

  // 3. copy past K/V into output cache regions
  copy_past_kernel<<<8192, 256, 0, stream>>>(past_k, past_v, Kout, Vout,
                                             2 * HEADS * PAST * HD / 4);

  // 4. QKV projections (WMMA)
  dim3 gg(DK / 128, MROWS / 128);
  gemm_wmma_kernel<<<gg, 256, 0, stream>>>(xb, Wtq, bq, nullptr, Qb, 0);
  gemm_wmma_kernel<<<gg, 256, 0, stream>>>(xb, Wtk, bk, Kout, nullptr, 1);
  gemm_wmma_kernel<<<gg, 256, 0, stream>>>(xb, Wtv, bv, Vout, nullptr, 1);

  // 5. flash attention (WMMA)
  dim3 ag(2048 / 128, HEADS, 2);
  attn_wmma_kernel<<<ag, 256, 0, stream>>>(Qb, Kout, Vout, Cb);

  // 6. output projection (WMMA)
  gemm_wmma_kernel<<<gg, 256, 0, stream>>>(Cb, Wto, bo, out, nullptr, 2);
}
